// GAU_15857019257041
// MI455X (gfx1250) — compile-verified, matrix operand order checked by reading, not run on hardware
//
#include <hip/hip_runtime.h>

// ---------------- CDNA5 wave32 WMMA types / helpers ----------------
typedef __attribute__((ext_vector_type(16))) __bf16 v16bf;
typedef __attribute__((ext_vector_type(8)))  __bf16 v8bf;
typedef __attribute__((ext_vector_type(8)))  float  v8f;
typedef __attribute__((ext_vector_type(4)))  unsigned u32x4;
typedef __attribute__((ext_vector_type(4)))  int      i32x4;
typedef __attribute__((ext_vector_type(8)))  int      i32x8;

#define NSEQ 8192
#define DIM  1024
#define HID  2048
#define QKD  200
#define QKP  224   // qk dim padded to 7*32 for the WMMA K loop
#define QKS  232   // LDS stride for q/k tiles (464B = 29*16B, conflict-free)
#define PAD72 72   // LDS stride for 64-wide K tiles (144B = 9*16B)

// Tensor Data Mover availability (ROCm7.2 clang-22 = 5 args; clang>=23 = 6)
#if __has_builtin(__builtin_amdgcn_tensor_load_to_lds) && \
    __has_builtin(__builtin_amdgcn_s_wait_tensorcnt)
#define USE_TDM 1
#else
#define USE_TDM 0
#endif

__device__ __forceinline__ v8f vzero() {
    v8f z;
#pragma unroll
    for (int i = 0; i < 8; ++i) z[i] = 0.f;
    return z;
}

__device__ __forceinline__ v8f wmma_bf16(v16bf a, v16bf b, v8f c) {
    return __builtin_amdgcn_wmma_f32_16x16x32_bf16(false, a, false, b,
                                                   (short)0, c, false, false);
}

// Fragment loader: element (r,k) at p[r*ld + k]; two ds_load_b128 per frag.
__device__ __forceinline__ v16bf frag_ld(const __bf16* p, int ld) {
    const int lane = threadIdx.x & 31;
    const int r  = lane & 15;
    const int kh = (lane >> 4) << 3;
    const __bf16* q = p + r * ld + kh;
    v8bf lo = *(const v8bf*)q;
    v8bf hi = *(const v8bf*)(q + 16);
    return __builtin_shufflevector(lo, hi, 0, 1, 2, 3, 4, 5, 6, 7,
                                   8, 9, 10, 11, 12, 13, 14, 15);
}

__device__ __forceinline__ unsigned pack_bf16(float a, float b) {
    union { __bf16 h[2]; unsigned u; } p;
    p.h[0] = (__bf16)a; p.h[1] = (__bf16)b;
    return p.u;
}

__device__ __forceinline__ float silu_f(float v) {
    return v / (1.f + __expf(-v));
}

// ---------------- Kernel 1: hid = silu(x@Wh + bh) -> v^T(bf16), gate(f32) ---
__global__ void __launch_bounds__(256)
k_hidden(const float* __restrict__ x, const float* __restrict__ Wh,
         const float* __restrict__ bh, __bf16* __restrict__ vt,  // [HID][NSEQ]
         float* __restrict__ gate) {
    extern __shared__ __bf16 smem[];
    __bf16* xs = smem;                    // 64 x 72 (A, row-major)
    __bf16* wt = smem + 64 * PAD72;       // 128(N) x 72(K)  (B transposed)
    const int tid  = threadIdx.x;
    const int wave = tid >> 5;
    const int m0   = blockIdx.x * 64;
    const int n0   = blockIdx.y * 128;
    const int msub = wave & 3;
    const int ngrp = wave >> 2;           // 0..1

    v8f acc[4];
#pragma unroll
    for (int j = 0; j < 4; ++j) acc[j] = vzero();

    unsigned* xs32 = (unsigned*)xs;
    unsigned* wt32 = (unsigned*)wt;
    for (int kc = 0; kc < DIM; kc += 64) {
#pragma unroll
        for (int i = 0; i < 8; ++i) {     // stage x: pack f32 pairs -> b32
            int idx = tid + i * 256, r = idx >> 5, c2 = idx & 31;
            float2 v = ((const float2*)(x + (size_t)(m0 + r) * DIM + kc))[c2];
            xs32[r * 36 + c2] = pack_bf16(v.x, v.y);
        }
#pragma unroll
        for (int i = 0; i < 16; ++i) {    // stage W transposed: [n][k] pairs
            int idx = tid + i * 256, n = idx & 127, rr = idx >> 7;
            float w0 = Wh[(size_t)(kc + 2 * rr)     * (2 * HID) + n0 + n];
            float w1 = Wh[(size_t)(kc + 2 * rr + 1) * (2 * HID) + n0 + n];
            wt32[n * 36 + rr] = pack_bf16(w0, w1);
        }
        __syncthreads();
#pragma unroll
        for (int ks = 0; ks < 64; ks += 32) {
            v16bf a = frag_ld(xs + msub * 16 * PAD72 + ks, PAD72);
#pragma unroll
            for (int j = 0; j < 4; ++j) {
                v16bf b = frag_ld(wt + (ngrp * 4 + j) * 16 * PAD72 + ks, PAD72);
                acc[j] = wmma_bf16(a, b, acc[j]);
            }
        }
        __syncthreads();
    }
    const int lane = tid & 31, nn = lane & 15, mb = (lane >> 4) << 3;
    const int rowb = m0 + msub * 16 + mb;
#pragma unroll
    for (int j = 0; j < 4; ++j) {
        int colg = n0 + (ngrp * 4 + j) * 16 + nn;
        if (colg < HID) {                 // v: pack 8 rows -> one b128 store (V^T)
            v8bf pv;
#pragma unroll
            for (int r = 0; r < 8; ++r)
                pv[r] = (__bf16)silu_f(acc[j][r] + bh[colg]);
            *(uint4*)(vt + (size_t)colg * NSEQ + rowb) =
                __builtin_bit_cast(uint4, pv);
        } else {                          // gate: f32 row-major
#pragma unroll
            for (int r = 0; r < 8; ++r) {
                float h = silu_f(acc[j][r] + bh[colg]);
                gate[(size_t)(rowb + r) * HID + (colg - HID)] = h;
            }
        }
    }
}

// ---------------- Kernel 2: qk = silu(x@Wqk + b); q,k affine (bf16, padded) -
__global__ void __launch_bounds__(256)
k_qk(const float* __restrict__ x, const float* __restrict__ Wqk,
     const float* __restrict__ bqk, const float* __restrict__ gamma,
     const float* __restrict__ beta, __bf16* __restrict__ qout,
     __bf16* __restrict__ kout) {
    extern __shared__ __bf16 smem[];
    __bf16* xs = smem;                    // 64 x 72
    __bf16* wt = smem + 64 * PAD72;       // 224(N) x 72(K) transposed
    const int tid  = threadIdx.x;
    const int wave = tid >> 5;
    const int m0   = blockIdx.x * 64;
    const int msub = wave & 3;
    const int ngrp = wave >> 2;

    v8f acc[7];
#pragma unroll
    for (int j = 0; j < 7; ++j) acc[j] = vzero();

    unsigned* xs32 = (unsigned*)xs;
    unsigned* wt32 = (unsigned*)wt;
    for (int kc = 0; kc < DIM; kc += 64) {
#pragma unroll
        for (int i = 0; i < 8; ++i) {
            int idx = tid + i * 256, r = idx >> 5, c2 = idx & 31;
            float2 v = ((const float2*)(x + (size_t)(m0 + r) * DIM + kc))[c2];
            xs32[r * 36 + c2] = pack_bf16(v.x, v.y);
        }
#pragma unroll
        for (int i = 0; i < 28; ++i) {    // 224*32 pairs
            int idx = tid + i * 256, n = idx % 224, rr = idx / 224;
            float w0 = 0.f, w1 = 0.f;
            if (n < QKD) {
                w0 = Wqk[(size_t)(kc + 2 * rr)     * QKD + n];
                w1 = Wqk[(size_t)(kc + 2 * rr + 1) * QKD + n];
            }
            wt32[n * 36 + rr] = pack_bf16(w0, w1);
        }
        __syncthreads();
#pragma unroll
        for (int ks = 0; ks < 64; ks += 32) {
            v16bf a = frag_ld(xs + msub * 16 * PAD72 + ks, PAD72);
#pragma unroll
            for (int j = 0; j < 7; ++j) {
                v16bf b = frag_ld(wt + (ngrp * 7 + j) * 16 * PAD72 + ks, PAD72);
                acc[j] = wmma_bf16(a, b, acc[j]);
            }
        }
        __syncthreads();
    }
    const int lane = tid & 31, nn = lane & 15, mb = (lane >> 4) << 3;
#pragma unroll
    for (int j = 0; j < 7; ++j) {
        int colg = (ngrp * 7 + j) * 16 + nn;
#pragma unroll
        for (int r = 0; r < 8; ++r) {
            int row = m0 + msub * 16 + mb + r;
            float qv = 0.f, kv = 0.f;
            if (colg < QKD) {
                float h = silu_f(acc[j][r] + bqk[colg]);
                qv = h * gamma[colg]       + beta[colg];
                kv = h * gamma[QKD + colg] + beta[QKD + colg];
            }
            qout[(size_t)row * QKP + colg] = (__bf16)qv;
            kout[(size_t)row * QKP + colg] = (__bf16)kv;
        }
    }
}

// ---------------- Kernel 3: flash-style relu^2 attention, gate fused --------
// Block = 16 query rows x full H=2048, K chunks of 64. LDS ~295KB of the
// CDNA5 320KB/WGP. V^T tile loaded by the Tensor Data Mover when available.
__device__ __forceinline__ v16bf frag_vt(const __bf16* vsT, int n0, int kk) {
    const int lane = threadIdx.x & 31;
    const int nn = lane & 15;
    const int kh = (lane >> 4) << 3;
    const int row = n0 + nn;
#if USE_TDM
    const int sw = 0;                       // TDM writes linear rows
#else
    const int sw = row & 7;                 // XOR-16B swizzled copy path
#endif
    const int c0 = ((kk + kh) >> 3) ^ sw;
    const int c1 = ((kk + kh + 16) >> 3) ^ sw;
    const v8bf* base = (const v8bf*)(vsT + row * 64);
    v8bf lo = base[c0];
    v8bf hi = base[c1];
    return __builtin_shufflevector(lo, hi, 0, 1, 2, 3, 4, 5, 6, 7,
                                   8, 9, 10, 11, 12, 13, 14, 15);
}

__global__ void __launch_bounds__(256)
k_attn(const __bf16* __restrict__ qg, const __bf16* __restrict__ kg,
       const __bf16* __restrict__ vtg,   // V^T [HID][NSEQ] bf16
       const float* __restrict__ gate, __bf16* __restrict__ po) {
    extern __shared__ __bf16 smem[];
    __bf16* qs  = smem;                              // 16 x 232
    __bf16* ks  = smem + 16 * QKS;                   // 64 x 232
    __bf16* vsT = smem + (16 + 64) * QKS;            // 2048 x 64
    __bf16* sim = vsT + HID * 64;                    // 16 x 72
    const int tid  = threadIdx.x;
    const int wave = tid >> 5;
    const int m0   = blockIdx.x * 16;
    const float scale = 0.03125f;                    // 1/sqrt(1024)

    {   // load Q tile once: 16 rows of 28 uint4 -> stride-29 rows
        const uint4* src = (const uint4*)(qg + (size_t)m0 * QKP);
        uint4* dst = (uint4*)qs;
        for (int idx = tid; idx < 448; idx += 256)
            dst[(idx / 28) * 29 + (idx % 28)] = src[idx];
    }
    v8f acc[16];
#pragma unroll
    for (int j = 0; j < 16; ++j) acc[j] = vzero();

#if USE_TDM
    // LDS byte offset of vsT within the dynamic LDS block (base of allocation)
    const unsigned vsT_lds = (unsigned)((16 + 64) * QKS * 2);
#endif

    for (int kb = 0; kb < NSEQ; kb += 64) {
#if USE_TDM
        if (wave == 0) {
            // D# group 0: count=1 | lds_addr | 57b global addr | type=2
            uint64_t ga = (uint64_t)(uintptr_t)vtg + (uint64_t)kb * 2;
            u32x4 g0;
            g0[0] = 1u;
            g0[1] = vsT_lds;
            g0[2] = (unsigned)ga;
            g0[3] = (unsigned)(ga >> 32) | (2u << 30);
            // D# group 1: data_size=2B; tensor 8192x2048 stride 8192;
            // tile 64(keys) x 2048(rows); no pad, no multicast, no iterate
            i32x8 g1;
            g1[0] = (1 << 16);                                   // data_size
            g1[1] = (int)((8192u & 0xFFFFu) << 16);              // dim0 lo16
            g1[2] = (int)((8192u >> 16) | ((2048u & 0xFFFFu) << 16));
            g1[3] = (int)((2048u >> 16) | (64u << 16));          // dim1 hi | tile0
            g1[4] = (int)2048;                                   // tile1 | tile2=0
            g1[5] = (int)8192;                                   // dim0_stride lo
            g1[6] = 0;
            g1[7] = 0;
            i32x4 z4 = {0, 0, 0, 0};
#if defined(__clang_major__) && __clang_major__ >= 23
            i32x8 z8 = {0, 0, 0, 0, 0, 0, 0, 0};
            __builtin_amdgcn_tensor_load_to_lds(g0, g1, z4, z4, z8, 0);
#else
            __builtin_amdgcn_tensor_load_to_lds(g0, g1, z4, z4, 0);
#endif
        }
#else
        {   // fallback: stage V^T chunk with XOR-16B swizzle
            const uint4* src = (const uint4*)vtg;
            uint4* dst = (uint4*)vsT;
            const int kbc = kb >> 3;
#pragma unroll
            for (int i = 0; i < 64; ++i) {
                int idx = tid + i * 256, n = idx >> 3, c = idx & 7;
                dst[n * 8 + (c ^ (n & 7))] = src[(size_t)n * 1024 + kbc + c];
            }
        }
#endif
        {   // stage K chunk: 64 rows x 28 uint4 -> stride-29 rows
            const uint4* src = (const uint4*)(kg + (size_t)kb * QKP);
            uint4* dst = (uint4*)ks;
#pragma unroll
            for (int i = 0; i < 7; ++i) {
                int idx = tid + i * 256;
                dst[(idx / 28) * 29 + (idx % 28)] = src[idx];
            }
        }
        if (kb + 64 < NSEQ && tid < 224)   // prefetch next K chunk
            __builtin_prefetch(kg + (size_t)(kb + 64) * QKP + tid * 64, 0, 0);
        __syncthreads();                   // K tile visible (TDM still flying)

        if (wave < 4) {                // sim tile 16x64: one 16x16 subtile/wave
            v8f s = vzero();
#pragma unroll
            for (int kk = 0; kk < QKP; kk += 32) {
                v16bf a = frag_ld(qs + kk, QKS);
                v16bf b = frag_ld(ks + wave * 16 * QKS + kk, QKS); // B(K^T)=A(K)
                s = wmma_bf16(a, b, s);
            }
            const int lane = tid & 31, nn = lane & 15, mb = (lane >> 4) << 3;
#pragma unroll
            for (int r = 0; r < 8; ++r) {
                float v = s[r] * scale;
                v = fmaxf(v, 0.f);
                v *= v;                               // relu^2
                sim[(mb + r) * PAD72 + wave * 16 + nn] = (__bf16)v;
            }
        }
#if USE_TDM
        if (wave == 0) __builtin_amdgcn_s_wait_tensorcnt(0);
#endif
        __syncthreads();                   // sim + V tile visible to all

        v16bf a0 = frag_ld(sim + 0,  PAD72);
        v16bf a1 = frag_ld(sim + 32, PAD72);
#pragma unroll
        for (int j = 0; j < 16; ++j) {
            int nsub = wave * 16 + j;
            v16bf b0 = frag_vt(vsT, nsub * 16, 0);
            v16bf b1 = frag_vt(vsT, nsub * 16, 32);
            acc[j] = wmma_bf16(a0, b0, acc[j]);
            acc[j] = wmma_bf16(a1, b1, acc[j]);
        }
        __syncthreads();                   // before next chunk overwrites LDS
    }
    const int lane = tid & 31, nn = lane & 15, mb = (lane >> 4) << 3;
#pragma unroll
    for (int j = 0; j < 16; ++j) {
        int col = (wave * 16 + j) * 16 + nn;
#pragma unroll
        for (int r = 0; r < 8; ++r) {
            int row = m0 + mb + r;
            float g = gate[(size_t)row * HID + col];
            po[(size_t)row * HID + col] = (__bf16)(acc[j][r] * g);
        }
    }
}

// ---------------- Kernel 4: out = (po@Wo + bo) * x --------------------------
__global__ void __launch_bounds__(256)
k_out(const __bf16* __restrict__ po, const float* __restrict__ Wo,
      const float* __restrict__ bo, const float* __restrict__ x,
      float* __restrict__ out) {
    extern __shared__ __bf16 smem[];
    __bf16* as = smem;                    // 64 x 72 (A bf16)
    __bf16* wt = smem + 64 * PAD72;       // 128(N) x 72(K) transposed
    const int tid  = threadIdx.x;
    const int wave = tid >> 5;
    const int m0   = blockIdx.x * 64;
    const int n0   = blockIdx.y * 128;
    const int msub = wave & 3;
    const int ngrp = wave >> 2;

    v8f acc[4];
#pragma unroll
    for (int j = 0; j < 4; ++j) acc[j] = vzero();

    unsigned* wt32 = (unsigned*)wt;
    for (int kc = 0; kc < HID; kc += 64) {
#pragma unroll
        for (int i = 0; i < 2; ++i) {     // stage A: 512 uint4, stride-9 rows
            int idx = tid + i * 256, r = idx >> 3, c = idx & 7;
            ((uint4*)as)[r * 9 + c] =
                ((const uint4*)(po + (size_t)(m0 + r) * HID + kc))[c];
        }
#pragma unroll
        for (int i = 0; i < 16; ++i) {    // stage W transposed
            int idx = tid + i * 256, n = idx & 127, rr = idx >> 7;
            float w0 = Wo[(size_t)(kc + 2 * rr)     * DIM + n0 + n];
            float w1 = Wo[(size_t)(kc + 2 * rr + 1) * DIM + n0 + n];
            wt32[n * 36 + rr] = pack_bf16(w0, w1);
        }
        __syncthreads();
#pragma unroll
        for (int kss = 0; kss < 64; kss += 32) {
            v16bf a = frag_ld(as + msub * 16 * PAD72 + kss, PAD72);
#pragma unroll
            for (int j = 0; j < 4; ++j) {
                v16bf b = frag_ld(wt + (ngrp * 4 + j) * 16 * PAD72 + kss, PAD72);
                acc[j] = wmma_bf16(a, b, acc[j]);
            }
        }
        __syncthreads();
    }
    const int lane = tid & 31, nn = lane & 15, mb = (lane >> 4) << 3;
#pragma unroll
    for (int j = 0; j < 4; ++j) {
        int colg = n0 + (ngrp * 4 + j) * 16 + nn;
#pragma unroll
        for (int r = 0; r < 8; ++r) {
            int row = m0 + msub * 16 + mb + r;
            float v = acc[j][r] + bo[colg];
            v *= x[(size_t)row * DIM + colg];
            out[(size_t)row * DIM + colg] = v;
        }
    }
}

// ---------------- Launch ----------------------------------------------------
extern "C" void kernel_launch(void* const* d_in, const int* in_sizes, int n_in,
                              void* d_out, int out_size, void* d_ws, size_t ws_size,
                              hipStream_t stream) {
    const float* x     = (const float*)d_in[0];
    const float* Wh    = (const float*)d_in[1];
    const float* bh    = (const float*)d_in[2];
    const float* Wqk   = (const float*)d_in[3];
    const float* bqk   = (const float*)d_in[4];
    const float* gamma = (const float*)d_in[5];
    const float* beta  = (const float*)d_in[6];
    const float* Wo    = (const float*)d_in[7];
    const float* bo    = (const float*)d_in[8];
    float* out = (float*)d_out;

    char* wsb = (char*)d_ws;
    size_t off = 0;
    auto take = [&](size_t bytes) -> char* {
        char* p = wsb + off;
        off = (off + bytes + 255) & ~(size_t)255;
        return p;
    };
    __bf16* vt   = (__bf16*)take((size_t)NSEQ * HID * 2);   // V^T (bf16)
    float*  gate = (float*) take((size_t)NSEQ * HID * 4);   // gate (f32)
    __bf16* q_bf = (__bf16*)take((size_t)NSEQ * QKP * 2);   // q padded (bf16)
    __bf16* k_bf = (__bf16*)take((size_t)NSEQ * QKP * 2);   // k padded (bf16)
    __bf16* po   = (__bf16*)take((size_t)NSEQ * HID * 2);   // (attn@v)*gate

    k_hidden<<<dim3(NSEQ / 64, (2 * HID) / 128), 256,
               (64 * PAD72 + 128 * PAD72) * 2, stream>>>(x, Wh, bh, vt, gate);
    k_qk<<<dim3(NSEQ / 64), 256,
           (64 * PAD72 + 224 * PAD72) * 2, stream>>>(x, Wqk, bqk, gamma, beta,
                                                     q_bf, k_bf);
    k_attn<<<dim3(NSEQ / 16), 256,
             ((16 + 64) * QKS + HID * 64 + 16 * PAD72) * 2, stream>>>(
        q_bf, k_bf, vt, gate, po);
    k_out<<<dim3(NSEQ / 64, DIM / 128), 256,
            (64 * PAD72 + 128 * PAD72) * 2, stream>>>(po, Wo, bo, x, out);
}
